// SinkhornOptimizer_v4_75436805587074
// MI455X (gfx1250) — matrix operand order repaired and checked by reading, not compiled, not run on hardware
//
#include <hip/hip_runtime.h>

// Sinkhorn on MI455X (gfx1250):
//  - 1 workgroup (512 thr = 16 waves -> 4 waves/SIMD for LDS-latency hiding) per batch matrix
//  - K (f16) and K^T (f16) resident in LDS (padded rows), ~267 KB of 320 KB
//  - mat-vecs via broadcast-A v_wmma_f32_16x16x32_f16 (f32 accumulate), 1 output block/wave
//  - final transport plan recomputed from X in f32

#define HEAD   256
#define NBATCH 512
#define NTHR   512   // 16 waves -> 4 waves/SIMD at 1 WG/WGP
#define PITCH  264   // 256 + 8 halfs pad -> 132 dwords/row, stride mod 64 == 4 (bank-friendly)
#define NITER  100

typedef __attribute__((ext_vector_type(16))) _Float16 v16h;
typedef __attribute__((ext_vector_type(8)))  _Float16 v8h;
typedef __attribute__((ext_vector_type(8)))  float    v8f;

__device__ __forceinline__ v16h hcat(v8h lo, v8h hi) {
  return __builtin_shufflevector(lo, hi, 0,1,2,3,4,5,6,7,8,9,10,11,12,13,14,15);
}

// A fragment (broadcast vector chunk): halves 0..7 at p, halves 8..15 at p+16
__device__ __forceinline__ v16h load_afrag(const _Float16* p) {
  return hcat(*(const v8h*)p, *(const v8h*)(p + 16));
}
// B fragment: 16 contiguous halves of one row
__device__ __forceinline__ v16h load_bfrag(const _Float16* p) {
  return hcat(*(const v8h*)p, *(const v8h*)(p + 8));
}

// y = vec^T * B  (B read row-wise: lane n reads Brows[row(n)][k..k+15]), dst = 1/y.
// A is the broadcast vector, so every M-row of D equals y -> read D[0] on lanes 0..15.
// One 16-wide output block per wave (wave = 0..15).
__device__ __forceinline__ void sink_matvec(const _Float16* __restrict__ Brows,
                                            const _Float16* __restrict__ vh_in,
                                            _Float16* __restrict__ vh_out,
                                            float*    __restrict__ vf_out,
                                            int lane, int wave)
{
  const int n    = lane & 15;
  const int aoff = (lane & 16) ? 16 : 0;   // A frag: lanes 0-15 K=0..15, lanes 16-31 K=16..31
  const int boff = (lane & 16) ? 16 : 0;   // B frag: lanes 0-15 K=0..15, lanes 16-31 K=16..31
  const int row  = wave * 16 + n;

  const _Float16* ap = vh_in + aoff;
  const _Float16* bp = Brows + row * PITCH + boff;

  v8f acc = {};

  // Double-buffered: issue next k-block's loads before this k-block's WMMA.
  v16h A[2], B[2];
  A[0] = load_bfrag(ap);
  B[0] = load_bfrag(bp);

#pragma unroll
  for (int kb = 0; kb < 8; ++kb) {
    const int cur = kb & 1;
    const int nxt = cur ^ 1;
    if (kb < 7) {
      A[nxt] = load_bfrag(ap + (kb + 1) * 32);
      B[nxt] = load_bfrag(bp + (kb + 1) * 32);
    }
    acc = __builtin_amdgcn_wmma_f32_16x16x32_f16(false, A[cur], false, B[cur],
                                                 (short)0, acc, false, false);
  }

  if (lane < 16) {
    // D[0] on lanes 0..15 is y[n] (all M-rows identical by construction)
    float r = 1.0f / acc[0];
    const int j = wave * 16 + lane;
    vf_out[j] = r;
    vh_out[j] = (_Float16)r;
  }
}

__global__ __launch_bounds__(NTHR, 1)
void sinkhorn_wmma_kernel(const float* __restrict__ X, float* __restrict__ Out)
{
  extern __shared__ char smem_raw[];
  _Float16* Kh  = (_Float16*)smem_raw;       // [256][PITCH]  rows of K
  _Float16* KTh = Kh  + HEAD * PITCH;        // [256][PITCH]  rows of K^T
  _Float16* uh  = KTh + HEAD * PITCH;        // [256] f16 u
  _Float16* vh  = uh  + HEAD;                // [256] f16 v
  float*    uf  = (float*)(vh + HEAD);       // [256] f32 u
  float*    vf  = uf + HEAD;                 // [256] f32 v

  const int tid  = threadIdx.x;
  const int lane = tid & 31;
  const int wave = tid >> 5;                 // 0..15
  const int col  = tid & (HEAD - 1);         // column handled in phases 1/3
  const int rh   = tid >> 8;                 // 0/1: even/odd rows
  const size_t base = (size_t)blockIdx.x * (HEAD * HEAD);

  // Phase 1: K = exp(-X/0.1) into LDS (both orientations), coalesced global reads.
  for (int r = 0; r < HEAD / 2; ++r) {
    const int row = 2 * r + rh;
    if (row + 16 < HEAD)
      __builtin_prefetch(&X[base + (size_t)(row + 16) * HEAD + col], 0, 0);
    float x  = X[base + (size_t)row * HEAD + col];
    float kv = __expf(x * -10.0f);
    _Float16 kh = (_Float16)kv;
    Kh [row * PITCH + col] = kh;   // row-major K
    KTh[col * PITCH + row] = kh;   // row-major K^T
  }
  if (tid < HEAD) {
    uh[tid] = (_Float16)(1.0f / HEAD);
    uf[tid] = 1.0f / HEAD;
  }
  __syncthreads();

  // Phase 2: 100 Sinkhorn iterations, entirely from LDS.
  for (int it = 0; it < NITER; ++it) {
    // v = 1 / (K^T u): B[k][n] = K[k][n] = KT[n][k]  -> rows of KT
    sink_matvec(KTh, uh, vh, vf, lane, wave);
    __syncthreads();
    // u = 1 / (K v):   B[j][i] = K[i][j]            -> rows of K
    sink_matvec(Kh, vh, uh, uf, lane, wave);
    __syncthreads();
  }

  // Phase 3: P = diag(u) K diag(v); recompute K in f32 from X for output precision.
  for (int r = 0; r < HEAD / 2; ++r) {
    const int row = 2 * r + rh;
    if (row + 16 < HEAD)
      __builtin_prefetch(&X[base + (size_t)(row + 16) * HEAD + col], 0, 0);
    const size_t idx = base + (size_t)row * HEAD + col;
    float x = X[idx];
    Out[idx] = uf[row] * __expf(x * -10.0f) * vf[col];
  }
}

extern "C" void kernel_launch(void* const* d_in, const int* in_sizes, int n_in,
                              void* d_out, int out_size, void* d_ws, size_t ws_size,
                              hipStream_t stream)
{
  (void)in_sizes; (void)n_in; (void)d_ws; (void)ws_size; (void)out_size;
  const float* X = (const float*)d_in[0];
  float* Out = (float*)d_out;

  // LDS: two padded f16 matrices + f16/f32 u,v vectors (~267 KB of 320 KB/WGP)
  const size_t lds_bytes = (size_t)2 * HEAD * PITCH * 2  // Kh + KTh (f16)
                         + (size_t)2 * HEAD * 2          // uh + vh  (f16)
                         + (size_t)2 * HEAD * 4;         // uf + vf  (f32)

  hipFuncSetAttribute((const void*)sinkhorn_wmma_kernel,
                      hipFuncAttributeMaxDynamicSharedMemorySize, (int)lds_bytes);

  sinkhorn_wmma_kernel<<<NBATCH, NTHR, lds_bytes, stream>>>(X, Out);
}
